// CausalMultiHeadSelfAttention_82463372083919
// MI455X (gfx1250) — matrix operand order repaired and checked by reading, not compiled, not run on hardware
//
#include <hip/hip_runtime.h>
#include <hip/hip_bf16.h>

// ---------------------------------------------------------------------------
// Causal MHA with RoPE for MI455X (gfx1250, wave32, WMMA).
//   B=2, T=2048, D_MODEL=1024, H=16, D_K=64.
// Reference quirk replicated: V = x @ Wq.T + bq (Wv/bv unused).
// - GEMMs: v_wmma_f32_16x16x32_f16, B tile staged via async global->LDS
//   (global_load_async_to_lds_b128 + s_wait_asynccnt), double-buffered,
//   shared by 8 waves per block (8x L2 traffic reduction on weights).
// - Attention: flash-style online softmax, scores never hit HBM; causal mask
//   VALU only on the diagonal block; P staged through LDS (s_wait_dscnt).
// ---------------------------------------------------------------------------

typedef _Float16 v16h __attribute__((ext_vector_type(16)));
typedef _Float16 v8h  __attribute__((ext_vector_type(8)));
typedef _Float16 v4h  __attribute__((ext_vector_type(4)));
typedef float    v8f  __attribute__((ext_vector_type(8)));

#define D_MODEL 1024
#define NUM_HEADS 16
#define D_K 64
#define T_SEQ 2048
#define B_SZ 2
#define ROWS (B_SZ * T_SEQ) // 4096

static __device__ __forceinline__ v8f wmma_f16(v16h a, v16h b, v8f c) {
  // D = A(16x32 f16) x B(32x16 f16) + C(16x16 f32)
  return __builtin_amdgcn_wmma_f32_16x16x32_f16(false, a, false, b, (short)0, c,
                                                false, false);
}

// 16-bit A-matrix 16x32 fragment: lane<16 holds K 0..15,
// lane>=16 holds K 16..31, row M = lane&15. rowk0 = &A[row][kk].
static __device__ __forceinline__ v16h load_a_frag(const _Float16* rowk0, int lh) {
  const int off = lh ? 16 : 0;
  v8h lo = *(const v8h*)(rowk0 + off);
  v8h hi = *(const v8h*)(rowk0 + 8 + off);
  return __builtin_shufflevector(lo, hi, 0, 1, 2, 3, 4, 5, 6, 7, 8, 9, 10, 11,
                                 12, 13, 14, 15);
}

// 16-bit B-matrix 32x16 fragment: lane holds column N = lane&15, 16 contiguous
// K values starting at (lane<16 ? 0 : 16). colk0 = &Bcol[k=kk] (contiguous K).
static __device__ __forceinline__ v16h load_b_frag(const _Float16* colk0, int lh) {
  return *(const v16h*)(colk0 + (lh ? 16 : 0));
}

// LDS byte offset of a __shared__ object (generic LDS addr = aperture|offset).
static __device__ __forceinline__ unsigned lds_off(const void* p) {
  return (unsigned)(unsigned long long)p;
}

// ---------------------------------------------------------------------------
__global__ void cvt_f32_to_f16_x4(const float4* __restrict__ in,
                                  v4h* __restrict__ out, int n4) {
  int i = blockIdx.x * blockDim.x + threadIdx.x;
  if (i < n4) {
    float4 f = in[i];
    v4h h;
    h.x = (_Float16)f.x; h.y = (_Float16)f.y;
    h.z = (_Float16)f.z; h.w = (_Float16)f.w;
    out[i] = h;
  }
}

// ---------------------------------------------------------------------------
// D[m][n] = sum_k A[m][k] * W[n][k] + bias[n]   (A: [4096,1024], W: [1024,1024])
// Block = 8 waves -> 256x64 tile; B tile (64x32 f16) async-staged to LDS,
// double-buffered, shared by all 8 waves. Grid: 16 m-blocks x 16 n-blocks.
template <bool STORE_HALF>
__global__ __launch_bounds__(256) void gemm_xwT_bias(
    const _Float16* __restrict__ A, const _Float16* __restrict__ W,
    const float* __restrict__ bias, void* __restrict__ D) {
  __shared__ __align__(32) _Float16 Bsh[2][64 * 32]; // 2 x 4KB

  const int tid = threadIdx.x;
  const int lane = tid & 31;
  const int lh = lane >> 4, nl = lane & 15;
  const int wv = tid >> 5; // 0..7
  const int mb = blockIdx.x >> 4, nb = blockIdx.x & 15;
  const int n0 = nb * 64;
  const int m0 = mb * 256 + wv * 32;

  // Each thread async-copies 8 halfs (16B) of the 64x32 B tile per K-step.
  const int srow = tid >> 2;        // N within tile, 0..63
  const int schunk = (tid & 3) * 8; // k-chunk (halfs)
  const unsigned long long wbase = (unsigned long long)W;
  const unsigned goff0 = (unsigned)(((n0 + srow) * D_MODEL + schunk) * 2);
  const unsigned ldsd0 = lds_off(&Bsh[0][srow * 32 + schunk]);
  const unsigned ldsd1 = lds_off(&Bsh[1][srow * 32 + schunk]);

  // prologue: stage kk=0 into buffer 0
  asm volatile("global_load_async_to_lds_b128 %0, %1, %2"
               :: "v"(ldsd0), "v"(goff0), "s"(wbase) : "memory");
  asm volatile("s_wait_asynccnt 0" ::: "memory");
  __syncthreads();

  v8f c[2][4] = {};
  int cur = 0;
  for (int kk = 0; kk < D_MODEL; kk += 32) {
    if (kk + 32 < D_MODEL) { // stage next K-step into the other buffer
      const unsigned dst = cur ? ldsd0 : ldsd1;
      const unsigned src = goff0 + (unsigned)((kk + 32) * 2);
      asm volatile("global_load_async_to_lds_b128 %0, %1, %2"
                   :: "v"(dst), "v"(src), "s"(wbase) : "memory");
    }
    __builtin_prefetch(A + (m0 + nl) * D_MODEL + kk + 32, 0, 0);
    const _Float16* Bbuf = &Bsh[cur][0];
    v16h a0 = load_a_frag(A + (m0 + nl) * D_MODEL + kk, lh);
    v16h a1 = load_a_frag(A + (m0 + 16 + nl) * D_MODEL + kk, lh);
#pragma unroll
    for (int nf = 0; nf < 4; ++nf) {
      v16h b = *(const v16h*)(Bbuf + (nf * 16 + nl) * 32 + lh * 16);
      c[0][nf] = wmma_f16(a0, b, c[0][nf]);
      c[1][nf] = wmma_f16(a1, b, c[1][nf]);
    }
    asm volatile("s_wait_asynccnt 0" ::: "memory"); // my slice landed in LDS
    __syncthreads();                                // everyone's slice visible
    cur ^= 1;
  }
#pragma unroll
  for (int mi = 0; mi < 2; ++mi) {
#pragma unroll
    for (int nf = 0; nf < 4; ++nf) {
      const int n = n0 + nf * 16 + nl;
      const float bv = bias[n];
#pragma unroll
      for (int r = 0; r < 8; ++r) {
        const int m = m0 + mi * 16 + r + 8 * lh; // C layout: row=vgpr(+8), col=lane
        const float v = c[mi][nf][r] + bv;
        if (STORE_HALF)
          ((_Float16*)D)[m * D_MODEL + n] = (_Float16)v;
        else
          ((float*)D)[m * D_MODEL + n] = v;
      }
    }
  }
}

// ---------------------------------------------------------------------------
// RoPE on Pq/Pk and repack: Qh/Kh = [b,h,t,d] f16 (rotated);
// Vt = [b,h,d,t] f16 (transposed, NOT rotated; V == Pq per reference).
__global__ void rope_pack(const _Float16* __restrict__ Pq,
                          const _Float16* __restrict__ Pk,
                          const int* __restrict__ pos,
                          _Float16* __restrict__ Qh, _Float16* __restrict__ Kh,
                          _Float16* __restrict__ Vt) {
  const int npair = ROWS * (D_MODEL / 2);
  int p = blockIdx.x * blockDim.x + threadIdx.x;
  if (p >= npair) return;
  const int row = p >> 9;       // b*T + t
  const int j0 = (p & 511) * 2; // even model column
  const int t = row & (T_SEQ - 1);
  const int b = row >> 11;
  const int h = j0 >> 6;
  const int d0 = j0 & 63; // even head-dim index == 2i

  const float x1 = (float)Pq[row * D_MODEL + j0];
  const float x2 = (float)Pq[row * D_MODEL + j0 + 1];
  const float k1 = (float)Pk[row * D_MODEL + j0];
  const float k2 = (float)Pk[row * D_MODEL + j0 + 1];

  const float fpos = (float)pos[t];
  // inv_freq = theta^(-d0/64), theta=1e4 ; ln(1e4)/64 = 0.14391156...
  const float ang = fpos * __expf(-0.143911568f * (float)d0);
  const float cs = __cosf(ang), sn = __sinf(ang);

  const int bh = b * NUM_HEADS + h;
  const int qidx = (bh * T_SEQ + t) * D_K + d0;
  Qh[qidx]     = (_Float16)(x1 * cs - x2 * sn);
  Qh[qidx + 1] = (_Float16)(x1 * sn + x2 * cs);
  Kh[qidx]     = (_Float16)(k1 * cs - k2 * sn);
  Kh[qidx + 1] = (_Float16)(k1 * sn + k2 * cs);

  const int vidx = (bh * D_K + d0) * T_SEQ + t;
  Vt[vidx]         = Pq[row * D_MODEL + j0]; // V = Q-projection (reference)
  Vt[vidx + T_SEQ] = Pq[row * D_MODEL + j0 + 1];
}

// ---------------------------------------------------------------------------
// Flash attention: one wave per (b,h, 16-query tile); 32-key blocks.
__global__ __launch_bounds__(128) void flash_attn(
    const _Float16* __restrict__ Qh, const _Float16* __restrict__ Kh,
    const _Float16* __restrict__ Vt, _Float16* __restrict__ heads) {
  __shared__ __align__(32) _Float16 Pbuf[4][16 * 32]; // per-wave 16x32 P tile

  const int lane = threadIdx.x & 31;
  const int lh = lane >> 4, nl = lane & 15;
  const int wwave = threadIdx.x >> 5;
  const int wave = blockIdx.x * 4 + wwave;
  const int bh = wave >> 7;         // 0..31
  const int q0 = (wave & 127) * 16; // query tile base

  const _Float16* Qbase = Qh + bh * T_SEQ * D_K;
  const _Float16* Kbase = Kh + bh * T_SEQ * D_K;
  const _Float16* Vbase = Vt + bh * D_K * T_SEQ;

  // Q A-fragments for d 0..31 and 32..63 (row = q0 + nl)
  const v16h aq0 = load_a_frag(Qbase + (q0 + nl) * D_K, lh);
  const v16h aq1 = load_a_frag(Qbase + (q0 + nl) * D_K + 32, lh);

  v8f oo[4] = {}; // O accum 16x64: frag nf covers d = nf*16 + lane
  float mi[8], li[8];
#pragma unroll
  for (int r = 0; r < 8; ++r) { mi[r] = -__builtin_inff(); li[r] = 0.0f; }

  _Float16* P = &Pbuf[wwave][0];
  const float scale = 0.125f; // 1/sqrt(64)
  const v8f vzero = {};

  for (int kb = 0; kb <= q0 + 15; kb += 32) { // causal: keys <= q0+15
    // speculative prefetch of the next key block (K rows and V columns)
    __builtin_prefetch(Kbase + (kb + 32 + nl) * D_K, 0, 0);
    __builtin_prefetch(Vbase + nl * T_SEQ + kb + 32, 0, 0);
    // --- scores: S = Q x K^T (two 16-key subtiles) ---
    v8f s0 = wmma_f16(aq0, load_b_frag(Kbase + (kb + nl) * D_K, lh), vzero);
    s0 = wmma_f16(aq1, load_b_frag(Kbase + (kb + nl) * D_K + 32, lh), s0);
    v8f s1 = wmma_f16(aq0, load_b_frag(Kbase + (kb + 16 + nl) * D_K, lh), vzero);
    s1 = wmma_f16(aq1, load_b_frag(Kbase + (kb + 16 + nl) * D_K + 32, lh), s1);

    // mask is only ever needed on the diagonal block (kb <= q0 always holds)
    const bool need_mask = (kb + 31 > q0);
    const int kc0 = kb + nl, kc1 = kb + 16 + nl;
    float alpha[8];
#pragma unroll
    for (int r = 0; r < 8; ++r) {
      const int qrow = q0 + r + 8 * lh;
      float v0 = s0[r] * scale;
      float v1 = s1[r] * scale;
      if (need_mask) {
        if (kc0 > qrow) v0 = -__builtin_inff();
        if (kc1 > qrow) v1 = -__builtin_inff();
      }
      // row max across the 16 lanes of this half
      float rm = fmaxf(v0, v1);
      rm = fmaxf(rm, __shfl_xor(rm, 1, 16));
      rm = fmaxf(rm, __shfl_xor(rm, 2, 16));
      rm = fmaxf(rm, __shfl_xor(rm, 4, 16));
      rm = fmaxf(rm, __shfl_xor(rm, 8, 16));
      const float mn = fmaxf(mi[r], rm);
      alpha[r] = __expf(mi[r] - mn);
      mi[r] = mn;
      const float p0 = __expf(v0 - mn);
      const float p1 = __expf(v1 - mn);
      float rs = p0 + p1;
      rs += __shfl_xor(rs, 1, 16);
      rs += __shfl_xor(rs, 2, 16);
      rs += __shfl_xor(rs, 4, 16);
      rs += __shfl_xor(rs, 8, 16);
      li[r] = li[r] * alpha[r] + rs;
      // stage P (f16) into LDS: row m = r + 8*lh, cols nl / 16+nl
      P[(r + 8 * lh) * 32 + nl] = (_Float16)p0;
      P[(r + 8 * lh) * 32 + 16 + nl] = (_Float16)p1;
    }
    // rescale O by alpha (per row r)
#pragma unroll
    for (int r = 0; r < 8; ++r) {
      oo[0][r] *= alpha[r]; oo[1][r] *= alpha[r];
      oo[2][r] *= alpha[r]; oo[3][r] *= alpha[r];
    }
    // same-wave LDS ordering: drain DS ops before re-reading P as A-fragment
    asm volatile("s_wait_dscnt 0" ::: "memory");
    const v16h ap = load_a_frag(P + nl * 32, lh);
    // --- O += P x V : B column = head-dim d (Vt keys contiguous) ---
#pragma unroll
    for (int nf = 0; nf < 4; ++nf) {
      v16h bv = load_b_frag(Vbase + (nf * 16 + nl) * T_SEQ + kb, lh);
      oo[nf] = wmma_f16(ap, bv, oo[nf]);
    }
  }

  // normalize and store heads[b, q, h*64 + d] (f16)
  const int b = bh >> 4, h = bh & 15;
#pragma unroll
  for (int r = 0; r < 8; ++r) {
    const float inv = 1.0f / li[r];
    oo[0][r] *= inv; oo[1][r] *= inv; oo[2][r] *= inv; oo[3][r] *= inv;
  }
#pragma unroll
  for (int nf = 0; nf < 4; ++nf) {
    const int d = nf * 16 + nl;
#pragma unroll
    for (int r = 0; r < 8; ++r) {
      const int q = q0 + r + 8 * lh;
      heads[(b * T_SEQ + q) * D_MODEL + h * D_K + d] = (_Float16)oo[nf][r];
    }
  }
}

// ---------------------------------------------------------------------------
extern "C" void kernel_launch(void* const* d_in, const int* in_sizes, int n_in,
                              void* d_out, int out_size, void* d_ws,
                              size_t ws_size, hipStream_t stream) {
  (void)in_sizes; (void)n_in; (void)out_size; (void)ws_size;
  const float* x  = (const float*)d_in[0];
  const int* pos  = (const int*)d_in[1];
  const float* Wq = (const float*)d_in[2];
  const float* bq = (const float*)d_in[3];
  const float* Wk = (const float*)d_in[4];
  const float* bk = (const float*)d_in[5];
  // d_in[6]/d_in[7] (Wv,bv) unused: reference computes V with Wq,bq.
  const float* Wo = (const float*)d_in[8];
  const float* bo = (const float*)d_in[9];
  float* out = (float*)d_out;

  char* ws = (char*)d_ws;
  size_t off = 0;
  auto take = [&](size_t bytes) {
    char* p = ws + off;
    off += (bytes + 255) & ~(size_t)255;
    return p;
  };
  _Float16* x16   = (_Float16*)take((size_t)ROWS * D_MODEL * 2);
  _Float16* Wq16  = (_Float16*)take((size_t)D_MODEL * D_MODEL * 2);
  _Float16* Wk16  = (_Float16*)take((size_t)D_MODEL * D_MODEL * 2);
  _Float16* Wo16  = (_Float16*)take((size_t)D_MODEL * D_MODEL * 2);
  _Float16* Pq    = (_Float16*)take((size_t)ROWS * D_MODEL * 2);
  _Float16* Pk    = (_Float16*)take((size_t)ROWS * D_MODEL * 2);
  _Float16* Qh    = (_Float16*)take((size_t)ROWS * D_MODEL * 2);
  _Float16* Kh    = (_Float16*)take((size_t)ROWS * D_MODEL * 2);
  _Float16* Vt    = (_Float16*)take((size_t)ROWS * D_MODEL * 2);
  _Float16* heads = (_Float16*)take((size_t)ROWS * D_MODEL * 2);

  const int nX4 = ROWS * D_MODEL / 4;      // 1,048,576
  const int nW4 = D_MODEL * D_MODEL / 4;   // 262,144
  cvt_f32_to_f16_x4<<<(nX4 + 255) / 256, 256, 0, stream>>>((const float4*)x, (v4h*)x16, nX4);
  cvt_f32_to_f16_x4<<<(nW4 + 255) / 256, 256, 0, stream>>>((const float4*)Wq, (v4h*)Wq16, nW4);
  cvt_f32_to_f16_x4<<<(nW4 + 255) / 256, 256, 0, stream>>>((const float4*)Wk, (v4h*)Wk16, nW4);
  cvt_f32_to_f16_x4<<<(nW4 + 255) / 256, 256, 0, stream>>>((const float4*)Wo, (v4h*)Wo16, nW4);

  // 256 blocks x 256 threads; block = 256x64 tile with shared async B staging
  gemm_xwT_bias<true><<<256, 256, 0, stream>>>(x16, Wq16, bq, (void*)Pq);
  gemm_xwT_bias<true><<<256, 256, 0, stream>>>(x16, Wk16, bk, (void*)Pk);

  const int npair = ROWS * (D_MODEL / 2); // 2,097,152
  rope_pack<<<(npair + 255) / 256, 256, 0, stream>>>(Pq, Pk, pos, Qh, Kh, Vt);

  // 32 (b,h) x 128 q-tiles = 4096 waves -> 1024 blocks x 128 threads
  flash_attn<<<1024, 128, 0, stream>>>(Qh, Kh, Vt, heads);

  gemm_xwT_bias<false><<<256, 256, 0, stream>>>(heads, Wo16, bo, (void*)out);
}
